// DepLabeledGCN_35029753266391
// MI455X (gfx1250) — matrix operand order repaired and checked by reading, not compiled, not run on hardware
//
#include <hip/hip_runtime.h>
#include <stdint.h>

#define D_DIM    512
#define N_NODES  128
#define B_BATCH  8
#define L_LABELS 48
#define LP       24   // labels per aggregation pass (48KB LDS fp32)

#define BM 128
#define BN 128
#define BK 32
#define KSPLIT 4
#define LDA 40        // LDS stride in halfs (80B rows -> 16B aligned chunks)

typedef __attribute__((ext_vector_type(16))) __bf16 v16bf;
typedef __attribute__((ext_vector_type(8)))  __bf16 v8bf;
typedef __attribute__((ext_vector_type(8)))  float  v8f;
typedef __attribute__((ext_vector_type(4)))  float  v4f;

// ---------------------------------------------------------------------------
// GCN weights: w[l,d,k] (fp32) -> Bn[d, l*512+k] (bf16, N-major, k contiguous)
// ---------------------------------------------------------------------------
__global__ void k_convert_w_gcn(const float* __restrict__ w, __bf16* __restrict__ Bn) {
    const int gid = blockIdx.x * blockDim.x + threadIdx.x;   // 48*512*64 threads
    const int row = gid >> 6;            // l*512 + d
    const int kc  = gid & 63;            // 8-k chunk
    const int l = row >> 9;
    const int d = row & 511;
    const float* src = w + (size_t)row * D_DIM + kc * 8;
    v4f f0 = *(const v4f*)(src);
    v4f f1 = *(const v4f*)(src + 4);
    v8bf o;
#pragma unroll
    for (int u = 0; u < 4; ++u) { o[u] = (__bf16)f0[u]; o[u + 4] = (__bf16)f1[u]; }
    *(v8bf*)(Bn + (size_t)d * (L_LABELS * D_DIM) + l * D_DIM + kc * 8) = o;
}

// ---------------------------------------------------------------------------
// Plain fp32 -> bf16 copy (mlp weights are already [d][k] N-major)
// ---------------------------------------------------------------------------
__global__ void k_convert_plain(const float* __restrict__ src, __bf16* __restrict__ dst) {
    const int gid = blockIdx.x * blockDim.x + threadIdx.x;
    v4f f0 = *(const v4f*)(src + (size_t)gid * 8);
    v4f f1 = *(const v4f*)(src + (size_t)gid * 8 + 4);
    v8bf o;
#pragma unroll
    for (int u = 0; u < 4; ++u) { o[u] = (__bf16)f0[u]; o[u + 4] = (__bf16)f1[u]; }
    *(v8bf*)(dst + (size_t)gid * 8) = o;
}

// ---------------------------------------------------------------------------
// denom[b*N+i] = 1 + sum_j adj[b,i,j]
// ---------------------------------------------------------------------------
__global__ void k_denom(const float* __restrict__ adj, float* __restrict__ denom) {
    const int row = blockIdx.x * blockDim.x + threadIdx.x;
    if (row >= B_BATCH * N_NODES) return;
    const float* a = adj + (size_t)row * N_NODES;
    float s = 1.0f;
    for (int j = 0; j < N_NODES; ++j) s += a[j];
    denom[row] = s;
}

// ---------------------------------------------------------------------------
// Label-bucketed aggregation:
//   S[b,i,l,k] = sum_j adj[b,i,j] * (dep[b,i,j]==l) * h[b,j,k]   (bf16 out)
// ---------------------------------------------------------------------------
__global__ __launch_bounds__(256) void k_aggregate(const float* __restrict__ h,
                                                   const float* __restrict__ adj,
                                                   const int*   __restrict__ dep,
                                                   __bf16*      __restrict__ S) {
    __shared__ float acc[LP * D_DIM];
    const int bi  = blockIdx.x;          // b*N + i
    const int b   = bi >> 7;             // N = 128
    const int tid = threadIdx.x;
    const float* arow = adj + (size_t)bi * N_NODES;
    const int*   drow = dep + (size_t)bi * N_NODES;

    for (int p = 0; p < 2; ++p) {
        const int l0 = p * LP;
        for (int u = tid; u < LP * D_DIM; u += 256) acc[u] = 0.0f;
        __syncthreads();
        for (int j = 0; j < N_NODES; ++j) {
            const float av  = arow[j];   // uniform across block
            const int   lab = drow[j];
            if (av != 0.0f && lab >= l0 && lab < l0 + LP) {
                const float* hj = h + (size_t)(b * N_NODES + j) * D_DIM;
                const int base  = (lab - l0) * D_DIM;
                acc[base + tid]       += av * hj[tid];
                acc[base + tid + 256] += av * hj[tid + 256];
            }
        }
        __syncthreads();
        __bf16* srow = S + ((size_t)bi * L_LABELS + l0) * D_DIM;
        for (int u = tid; u < LP * D_DIM; u += 256) srow[u] = (__bf16)acc[u];
        __syncthreads();
    }
}

// ---------------------------------------------------------------------------
// bf16 WMMA GEMM: C(M x Nn) = A(M x Kfull) * B(Nn x Kfull, N-MAJOR), one
// contiguous K slice of length Kslice per blockIdx.z (deterministic split-K).
// BM=BN=128, BK=32; 8 waves as 4(M) x 2(N); wave tile 32x64 (8 accums).
// Double-buffered LDS via global_load_async_to_lds_b128 + s_wait_asynccnt.
// partial!=nullptr: store raw fp32 partial tile; else fused epilogue.
// ---------------------------------------------------------------------------
__global__ __launch_bounds__(256) void k_gemm_bf16(
        const __bf16* __restrict__ A, const __bf16* __restrict__ Bn,
        int M, int Nn, int Kfull, int Kslice,
        const float* __restrict__ denom, const float* __restrict__ bias,
        int relu, float* __restrict__ outF, __bf16* __restrict__ outB,
        float* __restrict__ partial) {
    __shared__ __bf16 As[2][BM * LDA];   // 2 x 128 rows x 32 k  (20 KB)
    __shared__ __bf16 Bs[2][BN * LDA];   // 2 x 128 rows x 32 k  (20 KB)

    const int tid   = threadIdx.x;
    const int lane  = tid & 31;
    const int wave  = tid >> 5;
    const int warpM = wave >> 1;         // 0..3 -> 32 rows each
    const int warpN = wave & 1;          // 0..1 -> 64 cols each
    const int m0 = blockIdx.y * BM;
    const int n0 = blockIdx.x * BN;
    const int K0 = blockIdx.z * Kslice;

    // staging coordinates (each thread: 2 A chunks + 2 B chunks of 8 halfs)
    const int sr = tid >> 1;             // 0..127 (row for both tiles)
    const int sc = (tid & 1) * 16;       // 0 or 16 (k chunk base)

    const __bf16* gA = A  + (size_t)(m0 + sr) * Kfull + K0 + sc;
    const __bf16* gB = Bn + (size_t)(n0 + sr) * Kfull + K0 + sc;

    auto prefetch = [&](int buf, int kt) {
        const __bf16* pa = gA + kt;
        uint32_t la = (uint32_t)(uintptr_t)(&As[buf][sr * LDA + sc]);
        asm volatile("global_load_async_to_lds_b128 %0, %1, off" :: "v"(la), "v"(pa) : "memory");
        asm volatile("global_load_async_to_lds_b128 %0, %1, off" :: "v"(la + 16), "v"(pa + 8) : "memory");
        const __bf16* pb = gB + kt;
        uint32_t lb = (uint32_t)(uintptr_t)(&Bs[buf][sr * LDA + sc]);
        asm volatile("global_load_async_to_lds_b128 %0, %1, off" :: "v"(lb), "v"(pb) : "memory");
        asm volatile("global_load_async_to_lds_b128 %0, %1, off" :: "v"(lb + 16), "v"(pb + 8) : "memory");
    };

    v8f c[2][4];
#pragma unroll
    for (int s = 0; s < 2; ++s)
#pragma unroll
        for (int t = 0; t < 4; ++t) c[s][t] = (v8f){};

    auto compute = [&](int cur) {
        // A fragments: lanes 0-15 -> rows, K split {kb..kb+7, 16+kb..16+kb+7}
        const int kb = (lane >> 4) * 8;
        v16bf a[2];
#pragma unroll
        for (int s = 0; s < 2; ++s) {
            const int r = warpM * 32 + s * 16 + (lane & 15);
            v8bf lo = *(const v8bf*)(&As[cur][r * LDA + kb]);
            v8bf hi = *(const v8bf*)(&As[cur][r * LDA + 16 + kb]);
            a[s] = __builtin_shufflevector(lo, hi, 0,1,2,3,4,5,6,7,8,9,10,11,12,13,14,15);
        }
        // B fragments: lanes 0-15 -> cols, K split {kb..kb+7, 16+kb..16+kb+7}
        const int kb2 = (lane >> 4) * 8;
        v16bf bf[4];
#pragma unroll
        for (int t = 0; t < 4; ++t) {
            const int n = warpN * 64 + t * 16 + (lane & 15);
            v8bf lo = *(const v8bf*)(&Bs[cur][n * LDA + kb2]);
            v8bf hi = *(const v8bf*)(&Bs[cur][n * LDA + 16 + kb2]);
            bf[t] = __builtin_shufflevector(lo, hi, 0,1,2,3,4,5,6,7,8,9,10,11,12,13,14,15);
        }
#pragma unroll
        for (int s = 0; s < 2; ++s)
#pragma unroll
            for (int t = 0; t < 4; ++t)
                c[s][t] = __builtin_amdgcn_wmma_f32_16x16x32_bf16(
                    false, a[s], false, bf[t], (short)0, c[s][t], false, false);
    };

    const int T = Kslice / BK;
    prefetch(0, 0);
    for (int it = 0; it < T - 1; ++it) {         // steady state: branch-free waits
        prefetch((it & 1) ^ 1, (it + 1) * BK);
        asm volatile("s_wait_asynccnt 0x4" ::: "memory");
        __syncthreads();                         // tile `it&1` resident for all waves
        compute(it & 1);
        __syncthreads();                         // protect buffer before next prefetch
    }
    asm volatile("s_wait_asynccnt 0x0" ::: "memory");
    __syncthreads();
    compute((T - 1) & 1);

    // Epilogue. C layout: lanes 0-15: N=lane, M=v; lanes 16-31: N=lane-16, M=8+v.
#pragma unroll
    for (int s = 0; s < 2; ++s) {
#pragma unroll
        for (int t = 0; t < 4; ++t) {
            const int rbase = m0 + warpM * 32 + s * 16 + 8 * (lane >> 4);
            const int col   = n0 + warpN * 64 + t * 16 + (lane & 15);
#pragma unroll
            for (int v = 0; v < 8; ++v) {
                const int row = rbase + v;
                float x = c[s][t][v];
                if (partial) {
                    partial[((size_t)blockIdx.z * M + row) * Nn + col] = x;
                } else {
                    if (denom) x /= denom[row];
                    if (bias)  x += bias[col];
                    if (relu)  x = fmaxf(x, 0.0f);
                    if (outF) outF[(size_t)row * Nn + col] = x;
                    if (outB) outB[(size_t)row * Nn + col] = (__bf16)x;
                }
            }
        }
    }
}

// ---------------------------------------------------------------------------
// Split-K reduction + fused epilogue: sum KSPLIT partials, /denom, +bias, relu
// ---------------------------------------------------------------------------
__global__ void k_reduce_ep(const float* __restrict__ partial, int M, int Nn,
                            const float* __restrict__ denom, const float* __restrict__ bias,
                            int relu, float* __restrict__ outF, __bf16* __restrict__ outB) {
    const int gid  = blockIdx.x * blockDim.x + threadIdx.x;   // M*Nn/4 threads
    const size_t base = (size_t)gid * 4;
    const int row = (int)(base / Nn);
    const int col = (int)(base % Nn);
    v4f s = *(const v4f*)(partial + base);
#pragma unroll
    for (int z = 1; z < KSPLIT; ++z)
        s += *(const v4f*)(partial + (size_t)z * M * Nn + base);
    const float dn = denom ? denom[row] : 1.0f;
#pragma unroll
    for (int u = 0; u < 4; ++u) {
        float x = s[u] / dn;
        if (bias) x += bias[col + u];
        if (relu) x = fmaxf(x, 0.0f);
        if (outF) outF[base + u] = x;
        if (outB) outB[base + u] = (__bf16)x;
    }
}

// ---------------------------------------------------------------------------
extern "C" void kernel_launch(void* const* d_in, const int* in_sizes, int n_in,
                              void* d_out, int out_size, void* d_ws, size_t ws_size,
                              hipStream_t stream) {
    (void)in_sizes; (void)n_in; (void)out_size; (void)ws_size;
    const float* gcn_inputs = (const float*)d_in[0];
    /* d_in[1] word_seq_len: unused by the reference computation */
    const float* adj       = (const float*)d_in[2];
    const int*   dep       = (const int*)  d_in[3];
    const float* w_params  = (const float*)d_in[4];
    const float* mlp_w0    = (const float*)d_in[5];
    const float* mlp_b0    = (const float*)d_in[6];
    const float* mlp_w1    = (const float*)d_in[7];
    const float* mlp_b1    = (const float*)d_in[8];
    float* out = (float*)d_out;

    char* ws = (char*)d_ws;
    size_t off = 0;
    auto carve = [&](size_t bytes) -> void* {
        void* p = ws + off;
        off = (off + bytes + 255) & ~(size_t)255;
        return p;
    };
    const int M    = B_BATCH * N_NODES;        // 1024
    const int KBIG = L_LABELS * D_DIM;         // 24576

    __bf16* Wt   = (__bf16*)carve((size_t)L_LABELS * D_DIM * D_DIM * 2); // 25.2 MB (N-major)
    __bf16* Wt0  = (__bf16*)carve((size_t)D_DIM * D_DIM * 2);
    __bf16* Wt1  = (__bf16*)carve((size_t)D_DIM * D_DIM * 2);
    __bf16* S    = (__bf16*)carve((size_t)M * L_LABELS * D_DIM * 2);     // 50.3 MB
    float*  dnm  = (float*) carve((size_t)M * 4);
    float*  hbuf = (float*) carve((size_t)M * D_DIM * 4);
    __bf16* hb0  = (__bf16*)carve((size_t)M * D_DIM * 2);
    __bf16* hb1  = (__bf16*)carve((size_t)M * D_DIM * 2);
    float*  part = (float*) carve((size_t)KSPLIT * M * D_DIM * 4);       // 8 MB

    const dim3 gBig(D_DIM / BN, M / BM, KSPLIT);   // (4, 8, 4)
    const dim3 gMlp(D_DIM / BN, M / BM, 1);        // (4, 8, 1)
    const int  redGrid = (M * D_DIM / 4) / 256;    // 512

    // One-time weight conversions (reused by both GCN layers)
    k_convert_w_gcn<<<(L_LABELS * D_DIM * 64) / 256, 256, 0, stream>>>(w_params, Wt);
    k_convert_plain<<<(D_DIM * D_DIM / 8) / 256, 256, 0, stream>>>(mlp_w0, Wt0);
    k_convert_plain<<<(D_DIM * D_DIM / 8) / 256, 256, 0, stream>>>(mlp_w1, Wt1);
    k_denom<<<4, 256, 0, stream>>>(adj, dnm);

    // GCN layer 1: aggregate -> split-K WMMA GEMM -> reduce(+/denom, relu) fp32
    k_aggregate<<<M, 256, 0, stream>>>(gcn_inputs, adj, dep, S);
    k_gemm_bf16<<<gBig, 256, 0, stream>>>(S, Wt, M, D_DIM, KBIG, KBIG / KSPLIT,
                                          nullptr, nullptr, 0, nullptr, nullptr, part);
    k_reduce_ep<<<redGrid, 256, 0, stream>>>(part, M, D_DIM, dnm, nullptr, 1, hbuf, nullptr);
    // GCN layer 2: output only needed as bf16 (feeds MLP)
    k_aggregate<<<M, 256, 0, stream>>>(hbuf, adj, dep, S);
    k_gemm_bf16<<<gBig, 256, 0, stream>>>(S, Wt, M, D_DIM, KBIG, KBIG / KSPLIT,
                                          nullptr, nullptr, 0, nullptr, nullptr, part);
    k_reduce_ep<<<redGrid, 256, 0, stream>>>(part, M, D_DIM, dnm, nullptr, 1, nullptr, hb0);
    // MLP: relu(x @ W0^T + b0), relu(x @ W1^T + b1)  (single slice, fused epilogue)
    k_gemm_bf16<<<gMlp, 256, 0, stream>>>(hb0, Wt0, M, D_DIM, D_DIM, D_DIM,
                                          nullptr, mlp_b0, 1, nullptr, hb1, nullptr);
    k_gemm_bf16<<<gMlp, 256, 0, stream>>>(hb1, Wt1, M, D_DIM, D_DIM, D_DIM,
                                          nullptr, mlp_b1, 1, out, nullptr, nullptr);
}
